// SinkhornAttention_36670430773875
// MI455X (gfx1250) — matrix operand order repaired and checked by reading, not compiled, not run on hardware
//
#include <hip/hip_runtime.h>

typedef __attribute__((ext_vector_type(16))) _Float16     v16h;
typedef __attribute__((ext_vector_type(8)))  float        v8f;
typedef __attribute__((ext_vector_type(4)))  unsigned int v4u;
typedef __attribute__((ext_vector_type(8)))  int          v8i;
typedef __attribute__((ext_vector_type(4)))  int          v4i;

#define WMMA16(a,b,c) __builtin_amdgcn_wmma_f32_16x16x32_f16(false,(a),false,(b),(short)0,(c),false,false)

static __device__ __forceinline__ int laneid() { return threadIdx.x & 31; }

// ---------------------------------------------------------------------------
// TDM: 2D f16 tile (tile_d1 rows x tile_d0 elems, row stride stride0 elems)
// from global -> contiguous LDS. D# per CDNA5 ISA 8.3/8.4 (group0/group1),
// groups 2/3 zero (2D). 6-arg builtin form (amdgpu-toolchain / clang-23).
// Issued per-wave; caller gates to one wave + barrier.
// ---------------------------------------------------------------------------
static __device__ __forceinline__ void tdm_load_2d_f16(
    const _Float16* gsrc, void* lds_dst,
    unsigned tensor_d0, unsigned tensor_d1, unsigned stride0,
    unsigned tile_d0, unsigned tile_d1) {
  unsigned long long ga = (unsigned long long)(uintptr_t)gsrc;
  v4u g0;
  g0[0] = 1u;                                               // count=1, user mode
  g0[1] = (unsigned)(uintptr_t)lds_dst;                     // lds_addr
  g0[2] = (unsigned)(ga & 0xFFFFFFFFu);                     // global_addr[31:0]
  g0[3] = (unsigned)((ga >> 32) & 0x1FFFFFFu) | (2u << 30); // global_addr[56:32] | type=2
  v8i g1;
  g1[0] = (int)(1u << 16);                                  // data_size = 2 bytes
  g1[1] = (int)((tensor_d0 & 0xFFFFu) << 16);               // tensor_dim0[15:0]
  g1[2] = (int)((tensor_d0 >> 16) | ((tensor_d1 & 0xFFFFu) << 16));
  g1[3] = (int)((tensor_d1 >> 16) | (tile_d0 << 16));       // tile_dim0
  g1[4] = (int)tile_d1;                                     // tile_dim1 (tile_dim2=0)
  g1[5] = (int)stride0;                                     // tensor_dim0_stride[31:0]
  g1[6] = 0;
  g1[7] = 0;
  v4i z4 = {};
  v8i z8 = {};
  __builtin_amdgcn_tensor_load_to_lds(g0, g1, z4, z4, z8, 0);
}

// A fragment (16x32 f16) from row-major f16 source, leading dim ld.
static __device__ __forceinline__ v16h frag_a_h(const _Float16* A, int ld) {
  const int lane = laneid(), half = lane >> 4, row = lane & 15;
  v16h f;
  const _Float16* p = A + row * ld + half * 8;
#pragma unroll
  for (int j = 0; j < 8; ++j) f[j] = p[j];
#pragma unroll
  for (int j = 0; j < 8; ++j) f[8 + j] = p[16 + j];
  return f;
}

// A fragment from row-major f32 source (softmax weights), converting to f16.
static __device__ __forceinline__ v16h frag_a_f(const float* A, int ld) {
  const int lane = laneid(), half = lane >> 4, row = lane & 15;
  v16h f;
  const float* p = A + row * ld + half * 8;
#pragma unroll
  for (int j = 0; j < 8; ++j) f[j] = (_Float16)p[j];
#pragma unroll
  for (int j = 0; j < 8; ++j) f[8 + j] = (_Float16)p[16 + j];
  return f;
}

// B fragment (32x16, element (K,N)) from N-major storage Bt[n*ld + k].
static __device__ __forceinline__ v16h frag_b_t(const _Float16* Bt, int ld) {
  const int lane = laneid(), col = lane & 15, kb = (lane >> 4) * 8;
  v16h f;
  const _Float16* p = Bt + col * ld + kb;
#pragma unroll
  for (int j = 0; j < 8; ++j) f[j] = p[j];
#pragma unroll
  for (int j = 0; j < 8; ++j) f[8 + j] = p[16 + j];
  return f;
}

// ---------------------------------------------------------------------------
// One-shot converters: inputs f32->f16; weights f32 (K x N) -> f16 N-major.
// ---------------------------------------------------------------------------
__global__ __launch_bounds__(256) void cvt16_kernel(
    const float* __restrict__ src, _Float16* __restrict__ dst) {
  int i = (blockIdx.x * 256 + threadIdx.x) * 4;
  float4 f = *(const float4*)(src + i);
  dst[i + 0] = (_Float16)f.x;
  dst[i + 1] = (_Float16)f.y;
  dst[i + 2] = (_Float16)f.z;
  dst[i + 3] = (_Float16)f.w;
}

__global__ __launch_bounds__(256) void wtr_kernel(
    const float* __restrict__ W, _Float16* __restrict__ Wt) {
  int e = blockIdx.x * 256 + threadIdx.x;  // e = n*512 + k
  int n = e >> 9, k = e & 511;
  Wt[e] = (_Float16)W[(size_t)k * 512 + n];
}

// ---------------------------------------------------------------------------
// GEMM: O[16384x512] = A16[16384x512] * Wt16(N-major) + bias.
// TDM stages both tiles; 64 WMMAs/wave. OutT = f16 (qkv) or f32 (final out).
// ---------------------------------------------------------------------------
template <typename OutT>
__global__ __launch_bounds__(256) void gemm512_kernel(
    const _Float16* __restrict__ A, const _Float16* __restrict__ Wt,
    const float* __restrict__ bias, OutT* __restrict__ O) {
  __shared__ _Float16 sA[64 * 128];    // [M][K]
  __shared__ _Float16 sBt[128 * 128];  // [N][K]
  const int tid = threadIdx.x;
  const int m0 = blockIdx.x * 64;
  const int n0 = blockIdx.y * 128;
  const int wave = tid >> 5;
  const int msub = wave >> 1;
  const int nbase = (wave & 1) * 4;
  v8f acc[4] = {};

  for (int kc = 0; kc < 4; ++kc) {
    if (tid < 32) {
      tdm_load_2d_f16(A + (size_t)m0 * 512 + kc * 128, sA, 512, 16384, 512, 128, 64);
      tdm_load_2d_f16(Wt + (size_t)n0 * 512 + kc * 128, sBt, 512, 512, 512, 128, 128);
      __builtin_amdgcn_s_wait_tensorcnt((short)0);
    }
    __syncthreads();
#pragma unroll
    for (int ks = 0; ks < 4; ++ks) {
      v16h a = frag_a_h(&sA[(msub * 16) * 128 + ks * 32], 128);
#pragma unroll
      for (int t = 0; t < 4; ++t) {
        v16h b = frag_b_t(&sBt[((nbase + t) * 16) * 128 + ks * 32], 128);
        acc[t] = WMMA16(a, b, acc[t]);
      }
    }
    __syncthreads();
  }
  const int lane = laneid(), half = lane >> 4, col = lane & 15;
#pragma unroll
  for (int t = 0; t < 4; ++t) {
    int cg = n0 + (nbase + t) * 16 + col;
    float bb = bias[cg];
#pragma unroll
    for (int r = 0; r < 8; ++r) {
      int rg = m0 + msub * 16 + r + 8 * half;
      O[(size_t)rg * 512 + cg] = (OutT)(acc[t][r] + bb);
    }
  }
}

// ---------------------------------------------------------------------------
// Reductions over f16 k/v with f32 accumulation.
// ---------------------------------------------------------------------------
__global__ __launch_bounds__(256) void reduce_kernel(
    const _Float16* __restrict__ k, const _Float16* __restrict__ v,
    float* __restrict__ sumk, float* __restrict__ ksum, float* __restrict__ vsum) {
  int t = blockIdx.x * 256 + threadIdx.x;   // 0..131071
  int c = t & 511, i = (t >> 9) & 63, b = t >> 15;
  if (blockIdx.y == 0) {
    float s = 0.f;
    for (int j = 0; j < 64; ++j) s += (float)k[(size_t)(b * 4096 + i * 64 + j) * 512 + c];
    sumk[t] = s;                             // [b][nb][c]
  } else if (blockIdx.y == 1) {
    float s = 0.f;
    for (int j = 0; j < 64; ++j) s += (float)k[(size_t)(b * 4096 + j * 64 + i) * 512 + c];
    ksum[t] = s;                             // [b][s][c]
  } else {
    float s = 0.f;
    for (int j = 0; j < 64; ++j) s += (float)v[(size_t)(b * 4096 + j * 64 + i) * 512 + c];
    vsum[t] = s;
  }
}

// ---------------------------------------------------------------------------
// Sort head: S = sumk_h @ Wsort + bsort, softmax rows, p1 = row-sum.
// ---------------------------------------------------------------------------
__global__ __launch_bounds__(256) void sort_kernel(
    const float* __restrict__ sumk, const float* __restrict__ Wsort,
    const float* __restrict__ bsort, float* __restrict__ p1) {
  __shared__ _Float16 a16[64 * 64];   // [n][hd]
  __shared__ _Float16 w16t[64 * 64];  // [o][hd]
  __shared__ float S[64 * 64];
  const int tid = threadIdx.x;
  const int b = blockIdx.x >> 3, h = blockIdx.x & 7;
  for (int e = tid; e < 4096; e += 256) {
    int n = e >> 6, d = e & 63;
    a16[e] = (_Float16)sumk[(size_t)(b * 64 + n) * 512 + h * 64 + d];
    w16t[(e & 63) * 64 + (e >> 6)] = (_Float16)Wsort[e];  // Wsort[d][o], e=d*64+o
  }
  __syncthreads();
  const int wave = tid >> 5, lane = laneid(), half = lane >> 4, col = lane & 15;
#pragma unroll
  for (int t = 0; t < 2; ++t) {
    int tile = wave * 2 + t, msub = tile >> 2, nsub = tile & 3;
    v8f acc = {};
#pragma unroll
    for (int ks = 0; ks < 2; ++ks) {
      v16h a = frag_a_h(&a16[(msub * 16) * 64 + ks * 32], 64);
      v16h bb = frag_b_t(&w16t[(nsub * 16) * 64 + ks * 32], 64);
      acc = WMMA16(a, bb, acc);
    }
    float bv = bsort[nsub * 16 + col];
#pragma unroll
    for (int r = 0; r < 8; ++r)
      S[(msub * 16 + r + 8 * half) * 64 + nsub * 16 + col] = acc[r] + bv;
  }
  __syncthreads();
  if (tid < 64) {
    float mx = -3.0e38f;
    for (int j = 0; j < 64; ++j) mx = fmaxf(mx, S[tid * 64 + j]);
    float sum = 0.f;
    for (int j = 0; j < 64; ++j) sum += __expf(S[tid * 64 + j] - mx);
    float inv = 1.f / sum;
    float acc = 0.f;
    for (int j = 0; j < 64; ++j) acc += __expf(S[tid * 64 + j] - mx) * inv;
    p1[(size_t)b * 512 + tid * 8 + h] = acc;  // [b][n][h]
  }
}

// ---------------------------------------------------------------------------
// Block attention: one (b, block, h) per WG, two passes share accumulators.
// q/k tiles staged by TDM; v (transposed) and p1-scaled tiles staged manually.
// ---------------------------------------------------------------------------
static __device__ __forceinline__ void attn_pass(
    const _Float16* qh, const _Float16* kbuf, const _Float16* vbufT,
    float* S, v8f* acco) {
  const int tid = threadIdx.x, wave = tid >> 5;
  const int lane = tid & 31, half = lane >> 4, col = lane & 15;
#pragma unroll
  for (int t = 0; t < 2; ++t) {
    int tile = wave * 2 + t, msub = tile >> 2, nsub = tile & 3;
    v8f acc = {};
#pragma unroll
    for (int ks = 0; ks < 2; ++ks) {
      v16h a = frag_a_h(&qh[(msub * 16) * 64 + ks * 32], 64);
      v16h bb = frag_b_t(&kbuf[(nsub * 16) * 64 + ks * 32], 64);  // k is [s][hd]=[N][K]
      acc = WMMA16(a, bb, acc);
    }
#pragma unroll
    for (int r = 0; r < 8; ++r)
      S[(msub * 16 + r + 8 * half) * 64 + nsub * 16 + col] = acc[r] * 0.125f;
  }
  __syncthreads();
  if (tid < 64) {
    float mx = -3.0e38f;
    for (int j = 0; j < 64; ++j) mx = fmaxf(mx, S[tid * 64 + j]);
    float sum = 0.f;
    for (int j = 0; j < 64; ++j) sum += __expf(S[tid * 64 + j] - mx);
    float inv = 1.f / sum;
    for (int j = 0; j < 64; ++j) S[tid * 64 + j] = __expf(S[tid * 64 + j] - mx) * inv;
  }
  __syncthreads();
#pragma unroll
  for (int t = 0; t < 2; ++t) {
    int tile = wave * 2 + t, msub = tile >> 2, nsub = tile & 3;
#pragma unroll
    for (int ks = 0; ks < 2; ++ks) {
      v16h a = frag_a_f(&S[(msub * 16) * 64 + ks * 32], 64);
      v16h bb = frag_b_t(&vbufT[(nsub * 16) * 64 + ks * 32], 64);  // vT is [hd][s]=[N][K]
      acco[t] = WMMA16(a, bb, acco[t]);
    }
  }
  __syncthreads();
}

__global__ __launch_bounds__(256) void attn_kernel(
    const _Float16* __restrict__ q, const _Float16* __restrict__ k,
    const _Float16* __restrict__ v,
    const float* __restrict__ ksum, const float* __restrict__ vsum,
    const float* __restrict__ p1, _Float16* __restrict__ x) {
  __shared__ _Float16 qh[64 * 64], kh[64 * 64], vhT[64 * 64], k2[64 * 64], v2T[64 * 64];
  __shared__ float S[64 * 64];
  const int tid = threadIdx.x;
  const int h = blockIdx.x & 7, nb = (blockIdx.x >> 3) & 63, b = blockIdx.x >> 9;
  const int rowbase = b * 4096 + nb * 64;
  const float p1v = p1[(size_t)b * 512 + nb * 8 + h];
  if (tid < 32) {  // q,k tiles: straight 64x64 strided copies -> TDM
    tdm_load_2d_f16(q + (size_t)rowbase * 512 + h * 64, qh, 512, 16384, 512, 64, 64);
    tdm_load_2d_f16(k + (size_t)rowbase * 512 + h * 64, kh, 512, 16384, 512, 64, 64);
  }
  for (int e = tid; e < 4096; e += 256) {
    int s = e >> 6, d = e & 63;
    vhT[d * 64 + s] = v[(size_t)(rowbase + s) * 512 + h * 64 + d];
    size_t si = (size_t)(b * 64 + s) * 512 + h * 64 + d;
    k2[e] = (_Float16)(ksum[si] * p1v);            // sorted_key = p1 * ksum
    v2T[d * 64 + s] = (_Float16)(vsum[si] * p1v);
  }
  if (tid < 32) __builtin_amdgcn_s_wait_tensorcnt((short)0);
  __syncthreads();
  const int wave = tid >> 5, lane = tid & 31, half = lane >> 4, col = lane & 15;
  v8f acco[2] = {};
  attn_pass(qh, kh, vhT, S, acco);
  attn_pass(qh, k2, v2T, S, acco);
#pragma unroll
  for (int t = 0; t < 2; ++t) {
    int tile = wave * 2 + t, msub = tile >> 2, nsub = tile & 3;
#pragma unroll
    for (int r = 0; r < 8; ++r) {
      int s = msub * 16 + r + 8 * half;
      x[(size_t)(rowbase + s) * 512 + h * 64 + nsub * 16 + col] = (_Float16)acco[t][r];
    }
  }
}

// ---------------------------------------------------------------------------
// Launch. ws: a16,q16,k16,v16,x16 (5 x 16.8MB f16) + 4 Wt (2MB) + f32 tail.
// ---------------------------------------------------------------------------
extern "C" void kernel_launch(void* const* d_in, const int* in_sizes, int n_in,
                              void* d_out, int out_size, void* d_ws, size_t ws_size,
                              hipStream_t stream) {
  (void)in_sizes; (void)n_in; (void)out_size; (void)ws_size;
  const float* inq   = (const float*)d_in[0];
  const float* Wq    = (const float*)d_in[1];
  const float* bq    = (const float*)d_in[2];
  const float* Wk    = (const float*)d_in[3];
  const float* bk    = (const float*)d_in[4];
  const float* Wv    = (const float*)d_in[5];
  const float* bv    = (const float*)d_in[6];
  const float* Wsort = (const float*)d_in[7];
  const float* bsort = (const float*)d_in[8];
  const float* Wout  = (const float*)d_in[9];
  const float* bout  = (const float*)d_in[10];
  float* out = (float*)d_out;

  const size_t NROW = 16384, DIM = 512, MAT = NROW * DIM, WSZ = 512 * 512;
  _Float16* hbase = (_Float16*)d_ws;
  _Float16* a16 = hbase;
  _Float16* q16 = a16 + MAT;
  _Float16* k16 = q16 + MAT;
  _Float16* v16 = k16 + MAT;
  _Float16* x16 = v16 + MAT;
  _Float16* wqt = x16 + MAT;
  _Float16* wkt = wqt + WSZ;
  _Float16* wvt = wkt + WSZ;
  _Float16* wot = wvt + WSZ;
  float* fbase = (float*)(wot + WSZ);
  float* sumk = fbase;
  float* ksum = sumk + 4 * 64 * 512;
  float* vsum = ksum + 4 * 64 * 512;
  float* p1   = vsum + 4 * 64 * 512;

  dim3 blk(256);
  cvt16_kernel<<<dim3(MAT / 1024), blk, 0, stream>>>(inq, a16);
  wtr_kernel<<<dim3(1024), blk, 0, stream>>>(Wq, wqt);
  wtr_kernel<<<dim3(1024), blk, 0, stream>>>(Wk, wkt);
  wtr_kernel<<<dim3(1024), blk, 0, stream>>>(Wv, wvt);
  wtr_kernel<<<dim3(1024), blk, 0, stream>>>(Wout, wot);
  gemm512_kernel<_Float16><<<dim3(256, 4), blk, 0, stream>>>(a16, wqt, bq, q16);
  gemm512_kernel<_Float16><<<dim3(256, 4), blk, 0, stream>>>(a16, wkt, bk, k16);
  gemm512_kernel<_Float16><<<dim3(256, 4), blk, 0, stream>>>(a16, wvt, bv, v16);
  reduce_kernel<<<dim3(512, 3), blk, 0, stream>>>(k16, v16, sumk, ksum, vsum);
  sort_kernel<<<dim3(32), blk, 0, stream>>>(sumk, Wsort, bsort, p1);
  attn_kernel<<<dim3(2048), blk, 0, stream>>>(q16, k16, v16, ksum, vsum, p1, x16);
  gemm512_kernel<float><<<dim3(256, 4), blk, 0, stream>>>(x16, wot, bout, out);
}
